// GVPModel_455266533898
// MI455X (gfx1250) — matrix operand order repaired and checked by reading, not compiled, not run on hardware
//
#include <hip/hip_runtime.h>
#include <math.h>

// ---------------------------------------------------------------------------
// GVP-GNN forward for MI455X (gfx1250).
// Heavy GEMMs (edge messages m0/m1/m2, ff0/ff1, W_out, d0) run on
// v_wmma_f32_16x16x32_bf16 (bf16 inputs, f32 accumulate). Small vector-channel
// contractions (wh/wv), layernorms, gathers and segment-sum atomics stay VALU.
// GEMM contract: M multiple of 16; W padded to Np multiple of 64 (= NT*16) so
// the K-loop is straight-line: one A fragment feeds NT=4 back-to-back WMMAs
// on independent accumulators, no in-loop branches, no register shuffling.
// ---------------------------------------------------------------------------

#define NN 50000
#define EE 600000
#define GG 64
#define CHUNK 65536
#define NT 4            // N-tiles per wave in the GEMM (Np % (NT*16) == 0)

typedef __attribute__((ext_vector_type(16))) __bf16 v16bf;
typedef __attribute__((ext_vector_type(8)))  float  v8f;

struct BPack { uint4 lo, hi; };   // 32 bytes == v16bf

__device__ __forceinline__ unsigned short f2bfu(float x) {
  union { float f; unsigned u; } v; v.f = x;
  return (unsigned short)((v.u + 0x7FFFu + ((v.u >> 16) & 1u)) >> 16);  // RNE
}
__device__ __forceinline__ __bf16 bcvt(float x) {
  return __builtin_bit_cast(__bf16, f2bfu(x));
}
__device__ __forceinline__ float sigm(float x) { return 1.f / (1.f + __expf(-x)); }

// ---------------------------------------------------------------------------
// WMMA GEMM:  C[M,Nout] = act(A[M,Kp] * W[Np,Kp]^T + bias)
//   A fp32, K zero-padded to Kp (multiple of 32), M multiple of 16.
//   W bf16 [Np,Kp], Np multiple of 64 (zero-padded rows).
// 8 waves/block over M tiles; each wave owns exactly NT N-tiles.
// ---------------------------------------------------------------------------
__global__ void k_gemm_bf16_wmma(const float* __restrict__ A, int Kp,
                                 const unsigned short* __restrict__ Wp,
                                 const float* __restrict__ bias,
                                 float* __restrict__ C, int ldc,
                                 int M, int Nout, int relu) {
  const int lane  = threadIdx.x & 31;
  const int wid   = threadIdx.x >> 5;
  const int mtile = blockIdx.x * 8 + wid;
  const int Mt    = M >> 4;                 // M is a multiple of 16
  if (mtile >= Mt) return;                  // wave-uniform exit
  const int half  = lane >> 4;
  const int ml    = lane & 15;
  const int nbase = blockIdx.y * (NT * 16);

  const float*          Ap  = A  + (size_t)(mtile * 16 + ml) * Kp;
  const unsigned short* Wr0 = Wp + (size_t)(nbase + ml) * Kp;

  v8f acc[NT] = {};
  for (int k0 = 0; k0 < Kp; k0 += 32) {
    // ---- all fragment loads up front (one clause, one wait) ----
    // A fragment: lanes 0-15 take K [k0+0..7] & [k0+16..23]; lanes 16-31 +8.
    const float4* pa = (const float4*)(Ap + k0 + half * 8);
    const float4* qa = (const float4*)(Ap + k0 + 16 + half * 8);
    float4 f0 = pa[0], f1 = pa[1];
    float4 f2 = qa[0], f3 = qa[1];
    // B fragments: same K split as A: lanes 0-15 take K [k0+0..7] & [k0+16..23]; lanes 16-31 +8.
    BPack bp[NT];
#pragma unroll
    for (int nt = 0; nt < NT; ++nt) {
      const unsigned short* wb = Wr0 + (size_t)nt * 16 * Kp + k0 + half * 8;
      bp[nt].lo = *(const uint4*)wb; bp[nt].hi = *(const uint4*)(wb + 16);
    }
    // ---- pack A to bf16 ----
    v16bf a;
    a[0]  = bcvt(f0.x); a[1]  = bcvt(f0.y); a[2]  = bcvt(f0.z); a[3]  = bcvt(f0.w);
    a[4]  = bcvt(f1.x); a[5]  = bcvt(f1.y); a[6]  = bcvt(f1.z); a[7]  = bcvt(f1.w);
    a[8]  = bcvt(f2.x); a[9]  = bcvt(f2.y); a[10] = bcvt(f2.z); a[11] = bcvt(f2.w);
    a[12] = bcvt(f3.x); a[13] = bcvt(f3.y); a[14] = bcvt(f3.z); a[15] = bcvt(f3.w);
    // ---- NT back-to-back WMMAs on independent accumulators ----
#pragma unroll
    for (int nt = 0; nt < NT; ++nt) {
      v16bf bb = __builtin_bit_cast(v16bf, bp[nt]);
      acc[nt] = __builtin_amdgcn_wmma_f32_16x16x32_bf16(false, a, false, bb,
                                                        (short)0, acc[nt], false, false);
    }
  }

#pragma unroll
  for (int nt = 0; nt < NT; ++nt) {
    int col = nbase + nt * 16 + ml;
    if (col < Nout) {
      float bv = bias[col];
      float* Cp = C + (size_t)(mtile * 16 + half * 8) * ldc + col;
#pragma unroll
      for (int j = 0; j < 8; ++j) {         // C: lanes 16-31 hold rows m+8
        float v = acc[nt][j] + bv;
        if (relu) v = fmaxf(v, 0.f);
        Cp[(size_t)j * ldc] = v;
      }
    }
  }
}

// --------------------------- utility kernels -------------------------------
__global__ void k_zero(float* p, long n) {
  long t = (long)blockIdx.x * blockDim.x + threadIdx.x;
  if (t < n) p[t] = 0.f;
}
__global__ void k_zero_cols(float* A, int ld, int c0, int c1, long rows) {
  long w = c1 - c0;
  long t = (long)blockIdx.x * blockDim.x + threadIdx.x;
  if (t >= rows * w) return;
  A[(t / w) * ld + c0 + (t % w)] = 0.f;
}
__global__ void k_pack_w(const float* __restrict__ w, unsigned short* __restrict__ wp,
                         int No, int K, int Np, int Kp) {
  long t = (long)blockIdx.x * blockDim.x + threadIdx.x;
  if (t >= (long)Np * Kp) return;
  int r = (int)(t / Kp), k = (int)(t % Kp);
  wp[t] = (r < No && k < K) ? f2bfu(w[(size_t)r * K + k]) : (unsigned short)0;
}
// A[r, 0:sc) = S[r]; A[r, sc:ald) = 0   (vn columns filled later)
__global__ void k_copy_pad(float* __restrict__ A, int ald, const float* __restrict__ S,
                           int sld, int sc, long rows) {
  long t = (long)blockIdx.x * blockDim.x + threadIdx.x;
  if (t >= rows * ald) return;
  long r = t / ald; int k = (int)(t % ald);
  A[t] = (k < sc) ? S[r * sld + k] : 0.f;
}
// vh[r,h,3] = wh @ Vin[r,cin,3];  A[r, acol+h] = ||vh_h||
__global__ void k_gvp_vh(float* __restrict__ vh, float* __restrict__ A, int ald, int acol,
                         const float* __restrict__ Vin, const float* __restrict__ wh,
                         int cin, int h, long rows) {
  long t = (long)blockIdx.x * blockDim.x + threadIdx.x;
  if (t >= rows * h) return;
  long r = t / h; int hh = (int)(t % h);
  float a0 = 0.f, a1 = 0.f, a2 = 0.f;
  const float* vr = Vin + r * (size_t)cin * 3;
  for (int c = 0; c < cin; ++c) {
    float w = wh[hh * cin + c];
    a0 += vr[c * 3 + 0] * w; a1 += vr[c * 3 + 1] * w; a2 += vr[c * 3 + 2] * w;
  }
  float* o = vh + (r * h + hh) * 3;
  o[0] = a0; o[1] = a1; o[2] = a2;
  A[r * ald + acol + hh] = sqrtf(fmaxf(a0 * a0 + a1 * a1 + a2 * a2, 1e-8f));
}
// Vout[r,o,3] = wv @ vh[r,h,3] (optionally gated by sigmoid(||.||))
__global__ void k_gvp_vo(float* __restrict__ Vout, const float* __restrict__ vh,
                         const float* __restrict__ wv, int h, int o, long rows, int gate) {
  long t = (long)blockIdx.x * blockDim.x + threadIdx.x;
  if (t >= rows * o) return;
  long r = t / o; int oo = (int)(t % o);
  float a0 = 0.f, a1 = 0.f, a2 = 0.f;
  const float* vr = vh + r * (size_t)h * 3;
  for (int c = 0; c < h; ++c) {
    float w = wv[oo * h + c];
    a0 += vr[c * 3 + 0] * w; a1 += vr[c * 3 + 1] * w; a2 += vr[c * 3 + 2] * w;
  }
  if (gate) {
    float g = sigm(sqrtf(fmaxf(a0 * a0 + a1 * a1 + a2 * a2, 1e-8f)));
    a0 *= g; a1 *= g; a2 *= g;
  }
  float* p = Vout + (r * o + oo) * 3;
  p[0] = a0; p[1] = a1; p[2] = a2;
}

// ----------------------- model-specific kernels ----------------------------
__global__ void k_node_init(const float* __restrict__ node_s, const float* __restrict__ node_v,
                            const int* __restrict__ seq, const float* __restrict__ Ws,
                            const float* __restrict__ lnw, const float* __restrict__ lnb,
                            const float* __restrict__ wh, const float* __restrict__ wsw,
                            const float* __restrict__ wsb, const float* __restrict__ wv,
                            float* __restrict__ sN, float* __restrict__ vN, int N) {
  int n = blockIdx.x * blockDim.x + threadIdx.x;
  if (n >= N) return;
  float sc[42];
  for (int i = 0; i < 6; ++i) sc[i] = node_s[n * 6 + i];
  int sq = seq[n];
  for (int i = 0; i < 20; ++i) sc[6 + i] = Ws[sq * 20 + i];
  float mu = 0.f; for (int i = 0; i < 26; ++i) mu += sc[i]; mu /= 26.f;
  float var = 0.f; for (int i = 0; i < 26; ++i) { float d = sc[i] - mu; var += d * d; }
  float inv = rsqrtf(var / 26.f + 1e-5f);
  for (int i = 0; i < 26; ++i) sc[i] = (sc[i] - mu) * inv * lnw[i] + lnb[i];
  float vv[3][3]; float ss = 0.f;
  for (int c = 0; c < 3; ++c) {
    float s2 = 0.f;
    for (int d = 0; d < 3; ++d) { float x = node_v[n * 9 + c * 3 + d]; vv[c][d] = x; s2 += x * x; }
    ss += fmaxf(s2, 1e-8f);
  }
  float vsc = rsqrtf(ss / 3.f);
  for (int c = 0; c < 3; ++c) for (int d = 0; d < 3; ++d) vv[c][d] *= vsc;
  float vh[16][3];
  for (int h = 0; h < 16; ++h) {
    for (int d = 0; d < 3; ++d) {
      float a = 0.f;
      for (int c = 0; c < 3; ++c) a += vv[c][d] * wh[h * 3 + c];
      vh[h][d] = a;
    }
    sc[26 + h] = sqrtf(fmaxf(vh[h][0]*vh[h][0] + vh[h][1]*vh[h][1] + vh[h][2]*vh[h][2], 1e-8f));
  }
  for (int o = 0; o < 100; ++o) {
    float a = wsb[o];
    for (int k = 0; k < 42; ++k) a += sc[k] * wsw[o * 42 + k];
    sN[(size_t)n * 100 + o] = a;
  }
  for (int o = 0; o < 16; ++o)
    for (int d = 0; d < 3; ++d) {
      float a = 0.f;
      for (int h = 0; h < 16; ++h) a += vh[h][d] * wv[o * 16 + h];
      vN[(size_t)n * 48 + o * 3 + d] = a;
    }
}
__global__ void k_edge_init(const float* __restrict__ es_in, const float* __restrict__ ev_in,
                            const float* __restrict__ lnw, const float* __restrict__ lnb,
                            const float* __restrict__ wh, const float* __restrict__ wsw,
                            const float* __restrict__ wsb, const float* __restrict__ wv,
                            float* __restrict__ esH, float* __restrict__ evH, int E) {
  int e = blockIdx.x * blockDim.x + threadIdx.x;
  if (e >= E) return;
  float sc[33];
  for (int i = 0; i < 32; ++i) sc[i] = es_in[(size_t)e * 32 + i];
  float mu = 0.f; for (int i = 0; i < 32; ++i) mu += sc[i]; mu /= 32.f;
  float var = 0.f; for (int i = 0; i < 32; ++i) { float d = sc[i] - mu; var += d * d; }
  float inv = rsqrtf(var / 32.f + 1e-5f);
  for (int i = 0; i < 32; ++i) sc[i] = (sc[i] - mu) * inv * lnw[i] + lnb[i];
  float v0 = ev_in[(size_t)e * 3 + 0], v1 = ev_in[(size_t)e * 3 + 1], v2 = ev_in[(size_t)e * 3 + 2];
  float vsc = rsqrtf(fmaxf(v0 * v0 + v1 * v1 + v2 * v2, 1e-8f));
  v0 *= vsc; v1 *= vsc; v2 *= vsc;
  float w0 = wh[0];
  float h0 = w0 * v0, h1 = w0 * v1, h2 = w0 * v2;
  sc[32] = sqrtf(fmaxf(h0 * h0 + h1 * h1 + h2 * h2, 1e-8f));
  for (int o = 0; o < 32; ++o) {
    float a = wsb[o];
    for (int k = 0; k < 33; ++k) a += sc[k] * wsw[o * 33 + k];
    esH[(size_t)e * 32 + o] = a;
  }
  float wvv = wv[0];
  evH[(size_t)e * 3 + 0] = wvv * h0;
  evH[(size_t)e * 3 + 1] = wvv * h1;
  evH[(size_t)e * 3 + 2] = wvv * h2;
}
__global__ void k_count(float* cnt, const int* __restrict__ dst, int E) {
  int e = blockIdx.x * blockDim.x + threadIdx.x;
  if (e < E) atomicAdd(&cnt[dst[e]], 1.f);
}
// Pack A0 row: [s[src](100) | es(32) | s[dst](100) | (vn later) | 0-pad] -> 288
__global__ void k_conv_pack0(float* __restrict__ A0, const float* __restrict__ sN,
                             const float* __restrict__ esH, const int* __restrict__ src,
                             const int* __restrict__ dst, int e0, int ecnt) {
  long t = (long)blockIdx.x * blockDim.x + threadIdx.x;
  if (t >= (long)ecnt * 288) return;
  int e = (int)(t / 288), k = (int)(t % 288);
  int ge = e0 + e;
  float v = 0.f;
  if (k < 100)      v = sN[(size_t)src[ge] * 100 + k];
  else if (k < 132) v = esH[(size_t)ge * 32 + (k - 100)];
  else if (k < 232) v = sN[(size_t)dst[ge] * 100 + (k - 132)];
  A0[t] = v;
}
// m0 vector hidden: vh0 = wh0(33x33) @ concat(v[src],ev,v[dst]); vn -> A0[:,232+h]
__global__ void k_conv_vh0(float* __restrict__ vh0, float* __restrict__ A0,
                           const float* __restrict__ vN, const float* __restrict__ evH,
                           const int* __restrict__ src, const int* __restrict__ dst,
                           const float* __restrict__ wh, int e0, int ecnt) {
  long t = (long)blockIdx.x * blockDim.x + threadIdx.x;
  if (t >= (long)ecnt * 33) return;
  int e = (int)(t / 33), h = (int)(t % 33);
  int ge = e0 + e;
  int s = src[ge], dn = dst[ge];
  float a0 = 0.f, a1 = 0.f, a2 = 0.f;
  for (int c = 0; c < 16; ++c) {
    float w = wh[h * 33 + c];
    a0 += vN[(size_t)s * 48 + c * 3 + 0] * w;
    a1 += vN[(size_t)s * 48 + c * 3 + 1] * w;
    a2 += vN[(size_t)s * 48 + c * 3 + 2] * w;
  }
  {
    float w = wh[h * 33 + 16];
    a0 += evH[(size_t)ge * 3 + 0] * w;
    a1 += evH[(size_t)ge * 3 + 1] * w;
    a2 += evH[(size_t)ge * 3 + 2] * w;
  }
  for (int c = 0; c < 16; ++c) {
    float w = wh[h * 33 + 17 + c];
    a0 += vN[(size_t)dn * 48 + c * 3 + 0] * w;
    a1 += vN[(size_t)dn * 48 + c * 3 + 1] * w;
    a2 += vN[(size_t)dn * 48 + c * 3 + 2] * w;
  }
  float* o = vh0 + ((size_t)e * 33 + h) * 3;
  o[0] = a0; o[1] = a1; o[2] = a2;
  A0[(size_t)e * 288 + 232 + h] = sqrtf(fmaxf(a0 * a0 + a1 * a1 + a2 * a2, 1e-8f));
}
// Mean-aggregation scatter (segment_sum via f32 atomics)
__global__ void k_scatter(const float* __restrict__ ms, const float* __restrict__ mv,
                          const int* __restrict__ dst, float* __restrict__ asum,
                          float* __restrict__ avsum, int e0, int ecnt) {
  long t = (long)blockIdx.x * blockDim.x + threadIdx.x;
  if (t >= (long)ecnt * 148) return;
  int e = (int)(t / 148), k = (int)(t % 148);
  int d = dst[e0 + e];
  if (k < 100) atomicAdd(&asum[(size_t)d * 100 + k], ms[(size_t)e * 100 + k]);
  else         atomicAdd(&avsum[(size_t)d * 48 + (k - 100)], mv[(size_t)e * 48 + (k - 100)]);
}
// s,v = LayerNorm(s + ds/c, v + dv/c); c = max(cnt,1) or 1 if cnt==null
__global__ void k_residual_ln(float* __restrict__ s, float* __restrict__ v,
                              const float* __restrict__ ds, int dsld,
                              const float* __restrict__ dv, const float* __restrict__ cnt,
                              const float* __restrict__ w, const float* __restrict__ b, int N) {
  int n = blockIdx.x * blockDim.x + threadIdx.x;
  if (n >= N) return;
  float ic = cnt ? (1.f / fmaxf(cnt[n], 1.f)) : 1.f;
  size_t sb = (size_t)n * 100, db = (size_t)n * dsld, vb = (size_t)n * 48;
  float mu = 0.f;
  for (int k = 0; k < 100; ++k) mu += s[sb + k] + ds[db + k] * ic;
  mu *= 0.01f;
  float var = 0.f;
  for (int k = 0; k < 100; ++k) { float d = s[sb + k] + ds[db + k] * ic - mu; var += d * d; }
  float inv = rsqrtf(var * 0.01f + 1e-5f);
  for (int k = 0; k < 100; ++k) {
    float x = s[sb + k] + ds[db + k] * ic;
    s[sb + k] = (x - mu) * inv * w[k] + b[k];
  }
  float ss = 0.f;
  for (int c = 0; c < 16; ++c) {
    float s2 = 0.f;
    for (int d = 0; d < 3; ++d) { float x = v[vb + c * 3 + d] + dv[vb + c * 3 + d] * ic; s2 += x * x; }
    ss += fmaxf(s2, 1e-8f);
  }
  float vs = rsqrtf(ss * (1.f / 16.f));
  for (int j = 0; j < 48; ++j) { float x = v[vb + j] + dv[vb + j] * ic; v[vb + j] = x * vs; }
}
__global__ void k_store_outs(float* __restrict__ hs, float* __restrict__ hv,
                             const float* __restrict__ s, const float* __restrict__ v,
                             int L, int N) {
  long t = (long)blockIdx.x * blockDim.x + threadIdx.x;
  if (t >= (long)N * 148) return;
  int n = (int)(t / 148), k = (int)(t % 148);
  if (k < 100) hs[(size_t)n * 300 + L * 100 + k] = s[(size_t)n * 100 + k];
  else         hv[(size_t)n * 144 + L * 48 + (k - 100)] = v[(size_t)n * 48 + (k - 100)];
}
// ln_out: LN(300) -> Aout[:,0:300], v-norm -> hvln; Aout[:,348:352]=0
__global__ void k_final_ln(float* __restrict__ Aout, float* __restrict__ hvln,
                           const float* __restrict__ hs, const float* __restrict__ hv,
                           const float* __restrict__ w, const float* __restrict__ b, int N) {
  int n = blockIdx.x * blockDim.x + threadIdx.x;
  if (n >= N) return;
  size_t hb = (size_t)n * 300, ab = (size_t)n * 352, vb = (size_t)n * 144;
  float mu = 0.f; for (int k = 0; k < 300; ++k) mu += hs[hb + k]; mu /= 300.f;
  float var = 0.f; for (int k = 0; k < 300; ++k) { float d = hs[hb + k] - mu; var += d * d; }
  float inv = rsqrtf(var / 300.f + 1e-5f);
  for (int k = 0; k < 300; ++k) Aout[ab + k] = (hs[hb + k] - mu) * inv * w[k] + b[k];
  for (int k = 348; k < 352; ++k) Aout[ab + k] = 0.f;
  float ss = 0.f;
  for (int c = 0; c < 48; ++c) {
    float s2 = 0.f;
    for (int d = 0; d < 3; ++d) { float x = hv[vb + c * 3 + d]; s2 += x * x; }
    ss += fmaxf(s2, 1e-8f);
  }
  float vs = rsqrtf(ss * (1.f / 48.f));
  for (int j = 0; j < 144; ++j) hvln[vb + j] = hv[vb + j] * vs;
}
__global__ void k_head(const float* __restrict__ h600, const float* __restrict__ w,
                       const float* __restrict__ b, const int* __restrict__ gid,
                       float* __restrict__ out, float* __restrict__ gsum,
                       float* __restrict__ gcnt, int N) {
  int n = blockIdx.x * blockDim.x + threadIdx.x;
  if (n >= N) return;
  float a = 0.f;
  const float* hr = h600 + (size_t)n * 600;
  for (int k = 0; k < 600; ++k) a += hr[k] * w[k];
  a += b[0] + 0.5f;
  out[n] = a;
  int g = gid[n];
  atomicAdd(&gsum[g], a);
  atomicAdd(&gcnt[g], 1.f);
}
__global__ void k_gout(float* __restrict__ og, const float* __restrict__ gsum,
                       const float* __restrict__ gcnt, int G) {
  int g = blockIdx.x * blockDim.x + threadIdx.x;
  if (g < G) og[g] = gsum[g] / fmaxf(gcnt[g], 1.f);
}

// --------------------------------- host -----------------------------------
static inline unsigned blks(long tot) { return (unsigned)((tot + 255) / 256); }

static void gemm(hipStream_t st, const float* A, int Kp, const unsigned short* Wp,
                 const float* bias, float* C, int ldc, int M, int Np, int Nout, int relu) {
  int Mt = M / 16;                          // M multiple of 16 by contract
  int Ng = Np / (NT * 16);                  // Np multiple of 64 by contract
  dim3 grid((Mt + 7) / 8, Ng);
  k_gemm_bf16_wmma<<<grid, 256, 0, st>>>(A, Kp, Wp, bias, C, ldc, M, Nout, relu);
}

extern "C" void kernel_launch(void* const* d_in, const int* in_sizes, int n_in,
                              void* d_out, int out_size, void* d_ws, size_t ws_size,
                              hipStream_t stream) {
  (void)in_sizes; (void)out_size; (void)ws_size;
  const int N = NN, E = EE, G = GG, CH = CHUNK;

  const float* node_s = (const float*)d_in[0];
  const float* node_v = (const float*)d_in[1];
  const float* edge_s = (const float*)d_in[2];
  const float* edge_v = (const float*)d_in[3];
  const int*   seq    = (const int*)d_in[4];
  const int*   eidx   = (const int*)d_in[5];
  const int*   gid    = (const int*)d_in[6];
  const int* src = eidx;
  const int* dst = eidx + E;

  // ---- params: jax tree leaves in alphabetical nested-dict order ----------
  static const int layerSz[24] = {
    1089, 100, 26500, 528,   256, 100, 11600, 256,   256, 100, 11600, 256,
    512, 400, 52800, 1024,   1024, 100, 43200, 512,  100, 100, 100, 100 };
  static const int headSz[16] = { 1, 32, 1056, 1,  2304, 300, 104400,  400,
                                  48, 100, 4200, 256,  600, 180000, 1, 600 };
  static const int tailSz[6]  = { 32, 32, 300, 300, 26, 26 };
  int sizes[94]; int si = 0;
  for (int i = 0; i < 16; ++i) sizes[si++] = headSz[i];
  for (int L = 0; L < 3; ++L) for (int j = 0; j < 24; ++j) sizes[si++] = layerSz[j];
  for (int i = 0; i < 6; ++i) sizes[si++] = tailSz[i];

  const float* leaf[94];
  if (n_in >= 7 + 94) {
    for (int i = 0; i < 94; ++i) leaf[i] = (const float*)d_in[7 + i];
  } else {  // single flattened param blob
    const float* p = (const float*)d_in[7];
    size_t o = 0;
    for (int i = 0; i < 94; ++i) { leaf[i] = p + o; o += sizes[i]; }
  }

  // ---- workspace carve (16B-aligned so b128 fragment loads are legal) -----
  char* base = (char*)d_ws; size_t off = 0;
  auto AF = [&](size_t nf) -> float* {
    float* p = (float*)(base + off); off += (nf * 4 + 15) & ~(size_t)15; return p; };
  auto AB = [&](size_t nb) -> unsigned short* {
    unsigned short* p = (unsigned short*)(base + off);
    off += (nb * 2 + 15) & ~(size_t)15; return p; };

  float* sN    = AF((size_t)N * 100);
  float* vN    = AF((size_t)N * 48);
  float* esH   = AF((size_t)E * 32);
  float* evH   = AF((size_t)E * 3);
  float* hs    = AF((size_t)N * 300);
  float* hv    = AF((size_t)N * 144);
  float* asum  = AF((size_t)N * 100);
  float* avsum = AF((size_t)N * 48);
  float* cnt   = AF((size_t)N);
  float* gsum  = AF(G);
  float* gcnt  = AF(G);
  // W buffers: Np padded to multiple of 64 rows (zero-filled by k_pack_w)
  unsigned short *Wm0[3], *Wm1[3], *Wm2[3], *Wff0[3], *Wff1[3];
  for (int L = 0; L < 3; ++L) {
    Wm0[L]  = AB((size_t)128 * 288);
    Wm1[L]  = AB((size_t)128 * 128);
    Wm2[L]  = AB((size_t)128 * 128);
    Wff0[L] = AB((size_t)448 * 160);
    Wff1[L] = AB((size_t)128 * 448);
  }
  unsigned short* Wout = AB((size_t)320 * 352);
  unsigned short* Wd0  = AB((size_t)640 * 320);
  // Shared arena: conv chunk (1183*CH) / ff (948*N) / final (1560*N)
  size_t arenaF = (size_t)1560 * N;
  if ((size_t)1183 * CH > arenaF) arenaF = (size_t)1183 * CH;
  float* AR = AF(arenaF);
  // conv-phase view
  float* A0  = AR;
  float* vh0 = A0  + (size_t)CH * 288;
  float* ms0 = vh0 + (size_t)CH * 99;
  float* mv1 = ms0 + (size_t)CH * 100;
  float* vh1 = mv1 + (size_t)CH * 48;
  float* A1  = vh1 + (size_t)CH * 48;
  float* ms1 = A1  + (size_t)CH * 128;
  float* mv2 = ms1 + (size_t)CH * 100;
  float* vh2 = mv2 + (size_t)CH * 48;
  float* A2  = vh2 + (size_t)CH * 48;
  float* ms2 = A2  + (size_t)CH * 128;
  float* mv3 = ms2 + (size_t)CH * 100;
  // ff-phase view
  float* Aff0 = AR;
  float* vhf  = Aff0 + (size_t)N * 160;
  float* Aff1 = vhf  + (size_t)N * 96;
  float* vff  = Aff1 + (size_t)N * 448;
  float* dsb  = vff  + (size_t)N * 96;
  float* dvb  = dsb  + (size_t)N * 100;
  // final-phase view
  float* Aout = AR;
  float* vh48 = Aout + (size_t)N * 352;
  float* hvln = vh48 + (size_t)N * 144;
  float* Ad0  = hvln + (size_t)N * 144;
  float* h600 = Ad0  + (size_t)N * 320;

  // ---- pack GEMM weights to padded bf16 -----------------------------------
  auto packw = [&](const float* w, unsigned short* wp, int No, int K, int Np, int Kp) {
    k_pack_w<<<blks((long)Np * Kp), 256, 0, stream>>>(w, wp, No, K, Np, Kp); };
  for (int L = 0; L < 3; ++L) {
    int b = 16 + 24 * L;
    packw(leaf[b + 2],  Wm0[L],  100, 265, 128, 288);
    packw(leaf[b + 6],  Wm1[L],  100, 116, 128, 128);
    packw(leaf[b + 10], Wm2[L],  100, 116, 128, 128);
    packw(leaf[b + 14], Wff0[L], 400, 132, 448, 160);
    packw(leaf[b + 18], Wff1[L], 100, 432, 128, 448);
  }
  packw(leaf[6],  Wout, 300, 348, 320, 352);
  packw(leaf[13], Wd0,  600, 300, 640, 320);

  // ---- embeddings + degree counts -----------------------------------------
  k_zero<<<blks(N), 256, 0, stream>>>(cnt, N);
  k_count<<<blks(E), 256, 0, stream>>>(cnt, dst, E);
  k_node_init<<<blks(N), 256, 0, stream>>>(node_s, node_v, seq, leaf[7], leaf[93], leaf[92],
                                           leaf[8], leaf[10], leaf[9], leaf[11], sN, vN, N);
  k_edge_init<<<blks(E), 256, 0, stream>>>(edge_s, edge_v, leaf[89], leaf[88],
                                           leaf[0], leaf[2], leaf[1], leaf[3], esH, evH, E);

  // ---- message-passing layers ---------------------------------------------
  for (int L = 0; L < 3; ++L) {
    int b = 16 + 24 * L;
    k_zero<<<blks((long)N * 100), 256, 0, stream>>>(asum, (long)N * 100);
    k_zero<<<blks((long)N * 48), 256, 0, stream>>>(avsum, (long)N * 48);
    for (int e0 = 0; e0 < E; e0 += CH) {
      int ec = (E - e0 < CH) ? (E - e0) : CH;   // 65536 or 10176: both %16 == 0
      // m0
      k_conv_pack0<<<blks((long)ec * 288), 256, 0, stream>>>(A0, sN, esH, src, dst, e0, ec);
      k_conv_vh0<<<blks((long)ec * 33), 256, 0, stream>>>(vh0, A0, vN, evH, src, dst, leaf[b + 0], e0, ec);
      gemm(stream, A0, 288, Wm0[L], leaf[b + 1], ms0, 100, ec, 128, 100, 1);
      k_gvp_vo<<<blks((long)ec * 16), 256, 0, stream>>>(mv1, vh0, leaf[b + 3], 33, 16, ec, 1);
      // m1
      k_copy_pad<<<blks((long)ec * 128), 256, 0, stream>>>(A1, 128, ms0, 100, 100, ec);
      k_gvp_vh<<<blks((long)ec * 16), 256, 0, stream>>>(vh1, A1, 128, 100, mv1, leaf[b + 4], 16, 16, ec);
      gemm(stream, A1, 128, Wm1[L], leaf[b + 5], ms1, 100, ec, 128, 100, 1);
      k_gvp_vo<<<blks((long)ec * 16), 256, 0, stream>>>(mv2, vh1, leaf[b + 7], 16, 16, ec, 1);
      // m2
      k_copy_pad<<<blks((long)ec * 128), 256, 0, stream>>>(A2, 128, ms1, 100, 100, ec);
      k_gvp_vh<<<blks((long)ec * 16), 256, 0, stream>>>(vh2, A2, 128, 100, mv2, leaf[b + 8], 16, 16, ec);
      gemm(stream, A2, 128, Wm2[L], leaf[b + 9], ms2, 100, ec, 128, 100, 0);
      k_gvp_vo<<<blks((long)ec * 16), 256, 0, stream>>>(mv3, vh2, leaf[b + 11], 16, 16, ec, 0);
      // aggregate
      k_scatter<<<blks((long)ec * 148), 256, 0, stream>>>(ms2, mv3, dst, asum, avsum, e0, ec);
    }
    // residual + ln0 (mean over in-degree)
    k_residual_ln<<<blks(N), 256, 0, stream>>>(sN, vN, asum, 100, avsum, cnt,
                                               leaf[b + 21], leaf[b + 20], N);
    // ff0 -> ff1
    k_copy_pad<<<blks((long)N * 160), 256, 0, stream>>>(Aff0, 160, sN, 100, 100, N);
    k_gvp_vh<<<blks((long)N * 32), 256, 0, stream>>>(vhf, Aff0, 160, 100, vN, leaf[b + 12], 16, 32, N);
    k_zero_cols<<<blks((long)N * 16), 256, 0, stream>>>(Aff1, 448, 432, 448, N);
    gemm(stream, Aff0, 160, Wff0[L], leaf[b + 13], Aff1, 448, N, 448, 400, 1);
    k_gvp_vo<<<blks((long)N * 32), 256, 0, stream>>>(vff, vhf, leaf[b + 15], 32, 32, N, 1);
    k_gvp_vh<<<blks((long)N * 32), 256, 0, stream>>>(vhf, Aff1, 448, 400, vff, leaf[b + 16], 32, 32, N);
    gemm(stream, Aff1, 448, Wff1[L], leaf[b + 17], dsb, 100, N, 128, 100, 0);
    k_gvp_vo<<<blks((long)N * 16), 256, 0, stream>>>(dvb, vhf, leaf[b + 19], 32, 16, N, 0);
    // residual + ln1
    k_residual_ln<<<blks(N), 256, 0, stream>>>(sN, vN, dsb, 100, dvb, nullptr,
                                               leaf[b + 23], leaf[b + 22], N);
    k_store_outs<<<blks((long)N * 148), 256, 0, stream>>>(hs, hv, sN, vN, L, N);
  }

  // ---- readout ------------------------------------------------------------
  k_final_ln<<<blks(N), 256, 0, stream>>>(Aout, hvln, hs, hv, leaf[91], leaf[90], N);
  k_gvp_vh<<<blks((long)N * 48), 256, 0, stream>>>(vh48, Aout, 352, 300, hvln, leaf[4], 48, 48, N);
  k_zero_cols<<<blks((long)N * 20), 256, 0, stream>>>(Ad0, 320, 300, 320, N);
  gemm(stream, Aout, 352, Wout, leaf[5], Ad0, 320, N, 320, 300, 1);   // W_out (relu)
  gemm(stream, Ad0, 320, Wd0, leaf[12], h600, 600, N, 640, 600, 1);   // d0 (relu)
  k_zero<<<1, 256, 0, stream>>>(gsum, G);
  k_zero<<<1, 256, 0, stream>>>(gcnt, G);
  float* outp = (float*)d_out;
  k_head<<<blks(N), 256, 0, stream>>>(h600, leaf[15], leaf[14], gid, outp, gsum, gcnt, N);
  k_gout<<<1, 64, 0, stream>>>(outp + N, gsum, gcnt, G);
}
